// SConvLayer_46248207844050
// MI455X (gfx1250) — compile-verified, matrix operand order checked by reading, not run on hardware
//
#include <hip/hip_runtime.h>
#include <hip/hip_bf16.h>

typedef __attribute__((ext_vector_type(16))) _Float16 v16h;
typedef __attribute__((ext_vector_type(8)))  _Float16 v8h;
typedef __attribute__((ext_vector_type(8)))  float    v8f;
typedef __attribute__((ext_vector_type(4)))  unsigned int v4u;
typedef __attribute__((ext_vector_type(4)))  int      v4i;
typedef __attribute__((ext_vector_type(8)))  int      v8i;

#define D_DIM 512
#define H_DIM 8
#define I_DIM 64
#define L_DIM 1024
#define M_ROWS 2048   // B*L

#define BM 64
#define BN 128
#define BK 32
#define LDS_STRIDE 40   // halves; 80B rows keep 16B alignment of 8-half segments
#define XS_HALVES (BM * LDS_STRIDE)                      // 2560
#define SMEM_BYTES ((BM + BN) * LDS_STRIDE * 2)          // 15360

// TDM path: probe-guarded (safe fallback if builtins absent on this toolchain)
#if defined(__HIP_DEVICE_COMPILE__) && defined(__gfx1250__) && \
    __has_builtin(__builtin_amdgcn_tensor_load_to_lds) && \
    __has_builtin(__builtin_amdgcn_s_wait_tensorcnt)
#define USE_TDM 1
#else
#define USE_TDM 0
#endif

#if USE_TDM
// Issue one TENSOR_LOAD_TO_LDS: 2D f16 tile, tile_dim0 = BK (=32) contiguous
// elements, tileRows rows, row stride strideElems; LDS padded 4 DWORDs every
// 16 DWORDs (-> 40-half row stride, matching the compute-side layout).
__device__ __forceinline__ void tdm_load_tile_f16(
    const _Float16* gptr, unsigned ldsByteOff, int tileRows, int strideElems)
{
    unsigned long long ga = (unsigned long long)gptr;
    v4u g0;
    g0[0] = 1u;                                   // count=1, user descriptor
    g0[1] = ldsByteOff;                           // lds_addr
    g0[2] = (unsigned)(ga & 0xffffffffu);         // global_addr[31:0]
    g0[3] = (unsigned)(ga >> 32) | (2u << 30);    // global_addr[56:32] | type=2
    const unsigned td0 = 1u << 30, td1 = 1u << 30;   // huge dims: no OOB clipping
    v8i g1;
    g1[0] = (int)((1u << 16)      // data_size = 2 bytes
                | (1u << 20)      // pad_enable
                | (3u << 22)      // pad_interval: 16 DWORDs (64B = one 32-half row)
                | (3u << 25));    // pad_amount: 4 DWORDs (16B -> stride 40 halves)
    g1[1] = (int)((td0 & 0xffffu) << 16);                       // tensor_dim0 lo
    g1[2] = (int)((td0 >> 16) | ((td1 & 0xffffu) << 16));       // dim0 hi | dim1 lo
    g1[3] = (int)((td1 >> 16) | ((unsigned)BK << 16));          // dim1 hi | tile_dim0
    g1[4] = tileRows;                                           // tile_dim1 (dim2=0)
    g1[5] = strideElems;                                        // tensor_dim0_stride lo
    g1[6] = 0;
    g1[7] = 0;
    v4i gz = {0, 0, 0, 0};
#if __clang_major__ >= 23
    v8i gz8 = {0, 0, 0, 0, 0, 0, 0, 0};
    __builtin_amdgcn_tensor_load_to_lds(g0, g1, gz, gz, gz8, 0);
#else
    __builtin_amdgcn_tensor_load_to_lds(g0, g1, gz, gz, 0);
#endif
}
#endif

// ---------------------------------------------------------------------------
// Tiled WMMA GEMM (f16 operands, f32 accumulate):
//   Out[M,N] = X[M,K] @ W[K,N] + bias[N], optionally * mul[M,N]
// W is passed PRE-TRANSPOSED as WT[N,K] so both tiles stage contiguously.
// 256 threads (8 waves), 64x128 tile per WG, each wave a 32x32 sub-tile
// (2x2 v_wmma_f32_16x16x32_f16 with A/B fragment reuse).
// ---------------------------------------------------------------------------
__global__ __launch_bounds__(256) void gemm_wmma_f16(
    const _Float16* __restrict__ X, const _Float16* __restrict__ WT,
    const float* __restrict__ bias, const float* __restrict__ mul,
    float* __restrict__ Out, int N, int K)
{
    extern __shared__ __align__(16) _Float16 smem[];
    _Float16* Xs  = smem;                 // [BM][LDS_STRIDE], lds offset 0
    _Float16* WsT = smem + XS_HALVES;     // [BN][LDS_STRIDE], lds offset 5120 B

    const int tid  = threadIdx.x;
    const int lane = tid & 31;
    const int wave = tid >> 5;        // 0..7
    const int m_l  = lane & 15;       // row (A) / col (B,C/D) within 16
    const int g    = lane >> 4;       // lane group: K-half select / C row-half

    const int rowBase = blockIdx.y * BM;
    const int colBase = blockIdx.x * BN;
    const int wRow = (wave >> 2) * 32;   // 0 or 32
    const int wCol = (wave & 3) * 32;    // 0,32,64,96

    v8f acc[2][2] = {};

    for (int k0 = 0; k0 < K; k0 += BK) {
#if USE_TDM
        if (wave == 0) {   // one wave drives the Tensor Data Mover
            tdm_load_tile_f16(&X[(size_t)rowBase * K + k0],  0u,           BM, K);
            tdm_load_tile_f16(&WT[(size_t)colBase * K + k0], XS_HALVES * 2, BN, K);
            __builtin_amdgcn_s_wait_tensorcnt(0);
        }
        __syncthreads();
#else
        // ---- stage X tile (BM x BK): one b128 per thread ----
        {
            int r = tid >> 2;            // 0..63
            int q = (tid & 3) * 8;       // 0,8,16,24
            *(v8h*)&Xs[r * LDS_STRIDE + q] =
                *(const v8h*)&X[(size_t)(rowBase + r) * K + k0 + q];
        }
        // ---- stage WT tile (BN x BK): two b128 per thread ----
        {
            int r = tid >> 1;            // 0..127
            int q = (tid & 1) * 16;      // 0 or 16
            *(v8h*)&WsT[r * LDS_STRIDE + q] =
                *(const v8h*)&WT[(size_t)(colBase + r) * K + k0 + q];
            *(v8h*)&WsT[r * LDS_STRIDE + q + 8] =
                *(const v8h*)&WT[(size_t)(colBase + r) * K + k0 + q + 8];
        }
        if (k0 + BK < K) {  // CDNA5 global_prefetch_b8 on next K slice
            __builtin_prefetch(&X[(size_t)(rowBase + (tid >> 2)) * K + k0 + BK], 0, 0);
            __builtin_prefetch(&WT[(size_t)(colBase + (tid >> 1)) * K + k0 + BK], 0, 0);
        }
        __syncthreads();
#endif

        // ---- fragments per ISA 7.12.2: lane grp g holds K 8g..8g+7, 16+8g..16+8g+7
        v16h afr[2], bfr[2];
        #pragma unroll
        for (int mt = 0; mt < 2; ++mt) {
            const _Float16* row = &Xs[(wRow + mt * 16 + m_l) * LDS_STRIDE];
            v8h lo = *(const v8h*)&row[8 * g];
            v8h hi = *(const v8h*)&row[16 + 8 * g];
            afr[mt] = __builtin_shufflevector(lo, hi, 0,1,2,3,4,5,6,7,8,9,10,11,12,13,14,15);
        }
        #pragma unroll
        for (int nt = 0; nt < 2; ++nt) {
            const _Float16* row = &WsT[(wCol + nt * 16 + m_l) * LDS_STRIDE];
            v8h lo = *(const v8h*)&row[8 * g];
            v8h hi = *(const v8h*)&row[16 + 8 * g];
            bfr[nt] = __builtin_shufflevector(lo, hi, 0,1,2,3,4,5,6,7,8,9,10,11,12,13,14,15);
        }
        #pragma unroll
        for (int mt = 0; mt < 2; ++mt)
            #pragma unroll
            for (int nt = 0; nt < 2; ++nt)
                acc[mt][nt] = __builtin_amdgcn_wmma_f32_16x16x32_f16(
                    false, afr[mt], false, bfr[nt], (short)0, acc[mt][nt], false, false);
        __syncthreads();
    }

    // ---- epilogue: C/D layout VGPR r -> row (r + 8*g), col = lane%16 ----
    #pragma unroll
    for (int mt = 0; mt < 2; ++mt)
        #pragma unroll
        for (int nt = 0; nt < 2; ++nt)
            #pragma unroll
            for (int r = 0; r < 8; ++r) {
                int m = rowBase + wRow + mt * 16 + r + 8 * g;
                int n = colBase + wCol + nt * 16 + m_l;
                float v = acc[mt][nt][r] + bias[n];
                if (mul) v *= mul[(size_t)m * N + n];
                Out[(size_t)m * N + n] = v;
            }
}

// ---------------------------------------------------------------------------
__global__ void cvt_f16_kernel(const float* __restrict__ src,
                               _Float16* __restrict__ dst, int n)
{
    int i = blockIdx.x * blockDim.x + threadIdx.x;
    if (i < n) dst[i] = (_Float16)src[i];
}

// wT[n*K + k] = (f16) w[k*N + n] — LDS-tiled so both sides stay coalesced.
__global__ __launch_bounds__(256) void cvt_w_transpose(
    const float* __restrict__ w, _Float16* __restrict__ wT, int N, int K)
{
    __shared__ float tile[32][33];
    int kb = blockIdx.y * 32, nb = blockIdx.x * 32;
    int tx = threadIdx.x & 31, ty = threadIdx.x >> 5;   // 32 x 8
    #pragma unroll
    for (int j = 0; j < 32; j += 8)
        tile[ty + j][tx] = w[(size_t)(kb + ty + j) * N + nb + tx];
    __syncthreads();
    #pragma unroll
    for (int j = 0; j < 32; j += 8)
        wT[(size_t)(nb + ty + j) * K + kb + tx] = (_Float16)tile[tx][ty + j];
}

__global__ void silu_inplace(float* p, int n)
{
    int i = blockIdx.x * blockDim.x + threadIdx.x;
    if (i < n) { float v = p[i]; p[i] = v / (1.f + __expf(-v)); }
}

// ln_ph increments: lnI = silu(x@w_ang+b)*scale[h], lnR = -sigmoid(x@w_abs+b)*scale[h]
__global__ void smallproj(const float* __restrict__ x,
                          const float* __restrict__ w_ang, const float* __restrict__ b_ang,
                          const float* __restrict__ w_abs, const float* __restrict__ b_abs,
                          float* __restrict__ lnR, float* __restrict__ lnI)
{
    int idx = blockIdx.x * blockDim.x + threadIdx.x;  // row*8 + h
    if (idx >= M_ROWS * H_DIM) return;
    int row = idx >> 3, h = idx & 7;
    const float* xr = x + (size_t)row * D_DIM;
    float u = b_ang[h], v = b_abs[h];
    for (int k = 0; k < D_DIM; ++k) {
        float xv = xr[k];
        u += xv * w_ang[k * H_DIM + h];
        v += xv * w_abs[k * H_DIM + h];
    }
    float scale = __expf((float)h * (-6.907755278982137f / 7.0f)); // 0.001**(h/7)
    float sil = u / (1.f + __expf(-u));
    float sig = 1.f / (1.f + __expf(-v));
    lnI[idx] = sil * scale;
    lnR[idx] = -sig * scale;
}

// Inclusive scan along l for each (b,h): one wave32 per sequence.
__global__ __launch_bounds__(32) void scan_kernel(
    const float* __restrict__ lnR, const float* __restrict__ lnI,
    float* __restrict__ Sre, float* __restrict__ Sim)
{
    int b = blockIdx.x >> 3, h = blockIdx.x & 7;
    int lane = threadIdx.x;
    float cr = 0.f, ci = 0.f;
    for (int c = 0; c < L_DIM; c += 32) {
        int idx = (b * L_DIM + c + lane) * H_DIM + h;
        float vr = lnR[idx], vi = lnI[idx];
        #pragma unroll
        for (int off = 1; off < 32; off <<= 1) {
            float tr = __shfl_up(vr, off, 32);
            float ti = __shfl_up(vi, off, 32);
            if (lane >= off) { vr += tr; vi += ti; }
        }
        vr += cr; vi += ci;
        Sre[idx] = vr; Sim[idx] = vi;
        cr = __shfl(vr, 31, 32);
        ci = __shfl(vi, 31, 32);
    }
}

// ZS[b,h,i] = sum_{m>h} z[b,m,h,i]  (stored + lc_r),  Cbuf = z[b,h,h,i]
__global__ __launch_bounds__(64) void reduce_kernel(
    const float* __restrict__ zbuf, const float* __restrict__ lc_r,
    float* __restrict__ Abuf, float* __restrict__ Cbuf)
{
    int b = blockIdx.x >> 3, h = blockIdx.x & 7;
    int i = threadIdx.x;
    float s = 0.f;
    for (int m = h + 1; m < L_DIM; ++m)
        s += zbuf[(size_t)(b * L_DIM + m) * D_DIM + h * I_DIM + i];
    Abuf[(b * H_DIM + h) * I_DIM + i] = s + lc_r[h * I_DIM + i];
    Cbuf[(b * H_DIM + h) * I_DIM + i] = zbuf[(size_t)(b * L_DIM + h) * D_DIM + h * I_DIM + i];
}

// h = e^a (cos t * A - sin t * B) + C, then GroupNorm over I=64; emit f16
__global__ __launch_bounds__(64) void hnorm_kernel(
    const float* __restrict__ Sre, const float* __restrict__ Sim,
    const float* __restrict__ Abuf, const float* __restrict__ Cbuf,
    const float* __restrict__ lc_i,
    const float* __restrict__ gn_w, const float* __restrict__ gn_b,
    _Float16* __restrict__ hn16)
{
    __shared__ float red[64];
    int blk = blockIdx.x;           // row*8 + h
    int row = blk >> 3, h = blk & 7;
    int b = row >> 10;
    int i = threadIdx.x;

    float a = Sre[row * H_DIM + h], t = Sim[row * H_DIM + h];
    float e = __expf(a), cs = __cosf(t), sn = __sinf(t);
    float hv = e * (cs * Abuf[(b * H_DIM + h) * I_DIM + i] - sn * lc_i[h * I_DIM + i])
             + Cbuf[(b * H_DIM + h) * I_DIM + i];

    red[i] = hv; __syncthreads();
    #pragma unroll
    for (int s = 32; s > 0; s >>= 1) { if (i < s) red[i] += red[i + s]; __syncthreads(); }
    float mean = red[0] * (1.f / 64.f);
    __syncthreads();
    float d = hv - mean;
    red[i] = d * d; __syncthreads();
    #pragma unroll
    for (int s = 32; s > 0; s >>= 1) { if (i < s) red[i] += red[i + s]; __syncthreads(); }
    float var = red[0] * (1.f / 64.f);

    hn16[(size_t)row * D_DIM + h * I_DIM + i] =
        (_Float16)(d * rsqrtf(var + 1e-5f) * gn_w[h] + gn_b[h]);
}

// ---------------------------------------------------------------------------
extern "C" void kernel_launch(void* const* d_in, const int* in_sizes, int n_in,
                              void* d_out, int out_size, void* d_ws, size_t ws_size,
                              hipStream_t stream)
{
    (void)in_sizes; (void)n_in; (void)out_size; (void)ws_size;
    const float* x     = (const float*)d_in[0];
    const float* w_z   = (const float*)d_in[1];
    const float* b_z   = (const float*)d_in[2];
    const float* w_za  = (const float*)d_in[3];
    const float* b_za  = (const float*)d_in[4];
    const float* w_ang = (const float*)d_in[5];
    const float* b_ang = (const float*)d_in[6];
    const float* w_abs = (const float*)d_in[7];
    const float* b_abs = (const float*)d_in[8];
    const float* w_y   = (const float*)d_in[9];
    const float* b_y   = (const float*)d_in[10];
    const float* w_ya  = (const float*)d_in[11];
    const float* b_ya  = (const float*)d_in[12];
    const float* gn_w  = (const float*)d_in[13];
    const float* gn_b  = (const float*)d_in[14];
    const float* lc_r  = (const float*)d_in[15];
    const float* lc_i  = (const float*)d_in[16];
    float* out = (float*)d_out;

    const size_t MD = (size_t)M_ROWS * D_DIM;     // 1,048,576
    const size_t DD = (size_t)D_DIM * D_DIM;      //   262,144
    const size_t MH = (size_t)M_ROWS * H_DIM;     //    16,384

    char* base = (char*)d_ws;
    float* R0   = (float*)base;           base += MD * 4;   // z (gated value)
    float* R1   = (float*)base;           base += MD * 4;   // silu(x@w_za+b)
    float* R2   = (float*)base;           base += MD * 4;   // gate = silu(x@w_ya+b)
    float* lnR  = (float*)base;           base += MH * 4;
    float* lnI  = (float*)base;           base += MH * 4;
    float* Sre  = (float*)base;           base += MH * 4;
    float* Sim  = (float*)base;           base += MH * 4;
    float* Abuf = (float*)base;           base += 2 * H_DIM * I_DIM * 4;
    float* Cbuf = (float*)base;           base += 2 * H_DIM * I_DIM * 4;
    _Float16* x16    = (_Float16*)base;   base += MD * 2;
    _Float16* hn16   = (_Float16*)base;   base += MD * 2;
    _Float16* wzT16  = (_Float16*)base;   base += DD * 2;
    _Float16* wzaT16 = (_Float16*)base;   base += DD * 2;
    _Float16* wyaT16 = (_Float16*)base;   base += DD * 2;
    _Float16* wyT16  = (_Float16*)base;   base += DD * 2;

    // 0) one-time operand prep: x -> f16; weights -> f16 transposed [N,K]
    cvt_f16_kernel<<<(int)((MD + 255) / 256), 256, 0, stream>>>(x, x16, (int)MD);
    dim3 tgrid(D_DIM / 32, D_DIM / 32);   // (16,16)
    cvt_w_transpose<<<tgrid, 256, 0, stream>>>(w_z,  wzT16,  D_DIM, D_DIM);
    cvt_w_transpose<<<tgrid, 256, 0, stream>>>(w_za, wzaT16, D_DIM, D_DIM);
    cvt_w_transpose<<<tgrid, 256, 0, stream>>>(w_ya, wyaT16, D_DIM, D_DIM);
    cvt_w_transpose<<<tgrid, 256, 0, stream>>>(w_y,  wyT16,  D_DIM, D_DIM);

    dim3 ggrid(D_DIM / BN, M_ROWS / BM);   // (4, 32)

    // 1) R1 = silu(x @ w_za + b_za)
    gemm_wmma_f16<<<ggrid, 256, SMEM_BYTES, stream>>>(x16, wzaT16, b_za, nullptr, R1, D_DIM, D_DIM);
    silu_inplace<<<(int)((MD + 255) / 256), 256, 0, stream>>>(R1, (int)MD);
    // 2) R0 = (x @ w_z + b_z) * R1   (the gated value z)
    gemm_wmma_f16<<<ggrid, 256, SMEM_BYTES, stream>>>(x16, wzT16, b_z, R1, R0, D_DIM, D_DIM);
    // 3) R2 = silu(x @ w_ya + b_ya)  (output gate)
    gemm_wmma_f16<<<ggrid, 256, SMEM_BYTES, stream>>>(x16, wyaT16, b_ya, nullptr, R2, D_DIM, D_DIM);
    silu_inplace<<<(int)((MD + 255) / 256), 256, 0, stream>>>(R2, (int)MD);
    // 4) ln_ph increments
    smallproj<<<(int)((MH + 255) / 256), 256, 0, stream>>>(x, w_ang, b_ang, w_abs, b_abs, lnR, lnI);
    // 5) S = cumsum_l(ln_ph): 16 wave-level scans
    scan_kernel<<<2 * H_DIM, 32, 0, stream>>>(lnR, lnI, Sre, Sim);
    // 6) ZS + lc_r and diagonal z term
    reduce_kernel<<<2 * H_DIM, 64, 0, stream>>>(R0, lc_r, Abuf, Cbuf);
    // 7) complex epilogue + GroupNorm -> f16
    hnorm_kernel<<<M_ROWS * H_DIM, 64, 0, stream>>>(Sre, Sim, Abuf, Cbuf, lc_i, gn_w, gn_b, hn16);
    // 8) y = (hn @ w_y + b_y) * gate
    gemm_wmma_f16<<<ggrid, 256, SMEM_BYTES, stream>>>(hn16, wyT16, b_y, R2, out, D_DIM, D_DIM);
}
